// CrossNetMix_2216203125480
// MI455X (gfx1250) — matrix operand order repaired and checked by reading, not compiled, not run on hardware
//
#include <hip/hip_runtime.h>
#include <hip/hip_bf16.h>

typedef __attribute__((ext_vector_type(16))) __bf16 v16bf;
typedef __attribute__((ext_vector_type(8)))  __bf16 v8bf;
typedef __attribute__((ext_vector_type(8)))  float  v8f;

#define B_ 16384
#define D_ 1024
#define R_ 64
#define E_ 4
#define L_ 3

// Workspace layout in __bf16 elements: WMMA B-fragments, 32 lanes x 16 slots each.
#define WSU_OFF (L_*E_*4*32*32*16)            // V region size = 786432
#define WSC_OFF (WSU_OFF + L_*E_*64*2*32*16)  // + U region (786432) = 1572864
#define WS_TOT  (WSC_OFF + L_*E_*4*2*32*16)   // + C region (49152)  = 1622016

// ---------------------------------------------------------------------------
// Branchless tanh: hardware v_tanh_f32 on gfx1250 (confirmed in codegen).
// ---------------------------------------------------------------------------
__device__ __forceinline__ float fast_tanh(float x) {
#if __has_builtin(__builtin_amdgcn_tanhf)
  return __builtin_amdgcn_tanhf(x);
#elif __has_builtin(__builtin_amdgcn_tanh_f32)
  return __builtin_amdgcn_tanh_f32(x);
#else
  float e = __expf(2.f * x);
  return 1.f - 2.f * __builtin_amdgcn_rcpf(e + 1.f);
#endif
}

// ---------------------------------------------------------------------------
// Weight prep: fp32 -> bf16, packed into WMMA B-matrix fragment layout.
// B fragment (K x 16): lane l, slot j  ->  k = 16*(l/16) + j, n = l%16.
// ---------------------------------------------------------------------------
__global__ void prep_weights(const float* __restrict__ U,
                             const float* __restrict__ V,
                             const float* __restrict__ C,
                             __bf16* __restrict__ ws) {
  for (int idx = blockIdx.x * blockDim.x + threadIdx.x; idx < WS_TOT;
       idx += gridDim.x * blockDim.x) {
    float v;
    if (idx < WSU_OFF) {
      // V: B[k=d][n=r] for x @ V_e.  frag = ((le*4 + nt)*32 + kt)
      int t = idx; int j = t & 15; t >>= 4; int lane = t & 31; t >>= 5;
      int kt = t & 31; t >>= 5; int nt = t & 3; t >>= 2; int le = t;
      int k = ((lane >> 4) << 4) + j;
      int n = lane & 15;
      int d = kt * 32 + k, r = nt * 16 + n;
      v = V[((size_t)le * D_ + d) * R_ + r];
    } else if (idx < WSC_OFF) {
      // U^T: B[k=r][n=d] for v2 @ U_e^T.  frag = ((le*64 + nt)*2 + kt)
      int t = idx - WSU_OFF; int j = t & 15; t >>= 4; int lane = t & 31; t >>= 5;
      int kt = t & 1; t >>= 1; int nt = t & 63; t >>= 6; int le = t;
      int k = ((lane >> 4) << 4) + j;
      int n = lane & 15;
      int r = kt * 32 + k, d = nt * 16 + n;
      v = U[((size_t)le * D_ + d) * R_ + r];
    } else {
      // C^T: B[k=s][n=r_out] for v1 @ C_e^T.  frag = ((le*4 + nt)*2 + kt)
      int t = idx - WSC_OFF; int j = t & 15; t >>= 4; int lane = t & 31; t >>= 5;
      int kt = t & 1; t >>= 1; int nt = t & 3; t >>= 2; int le = t;
      int k = ((lane >> 4) << 4) + j;
      int n = lane & 15;
      int s = kt * 32 + k, ro = nt * 16 + n;
      v = C[((size_t)le * R_ + ro) * R_ + s];
    }
    ws[idx] = (__bf16)v;
  }
}

// A-fragment map (16x32 bf16 A), this kernel's own order, not the ISA table:
//   lane-half khi = K>>4,  slot j = K&15   (contiguous k order)
__device__ __forceinline__ void a_slot(int K, int& khi, int& j) {
  khi = K >> 4;
  j   = K & 15;
}

// ---------------------------------------------------------------------------
// Fully fused 3-layer CrossNetMix: one 16-row tile per workgroup.
// ---------------------------------------------------------------------------
__launch_bounds__(256)
__global__ void crossnet_kernel(const float* __restrict__ x0g,
                                const float* __restrict__ G,
                                const float* __restrict__ bias,
                                const __bf16* __restrict__ ws,
                                float* __restrict__ out) {
  __shared__ __align__(16) float  sX[16 * 1028];     // fp32 residual stream (padded rows)
  __shared__ __align__(64) __bf16 sA[32 * 512];      // bf16 x tile, A-fragment layout
  __shared__ __align__(64) __bf16 sV1[E_ * 1024];    // tanh(xV) per expert, A-frag layout
  __shared__ __align__(64) __bf16 sV2[E_ * 1024];    // tanh(Cv) per expert, A-frag layout
  __shared__ float sLogit[64];
  __shared__ float sGate[64];

  const int tid  = threadIdx.x;
  // Wave id as an explicit SGPR: task loops become scalar (no EXEC masking).
  const int w    = __builtin_amdgcn_readfirstlane(tid >> 5);
  const int lane = tid & 31;
  const size_t row0 = (size_t)blockIdx.x * 16;

  const v16bf* wsV = (const v16bf*)(ws);
  const v16bf* wsU = (const v16bf*)(ws + WSU_OFF);
  const v16bf* wsC = (const v16bf*)(ws + WSC_OFF);

  // Load the tile's rows of x0 into the fp32 residual buffer (float4, 16 fixed iters).
  {
    const float4* src = (const float4*)(x0g + row0 * D_);
    #pragma unroll 4
    for (int it = 0; it < 16; ++it) {
      int idx = tid + it * 256;
      int m = idx >> 8, c = idx & 255;
      *(float4*)&sX[m * 1028 + c * 4] = src[m * 256 + c];
    }
  }
  __syncthreads();

  for (int l = 0; l < L_; ++l) {
    // --- repack x_l to bf16 A-fragments (8 consecutive d = one b128 store) ---
    if (tid < 64) sLogit[tid] = 0.f;
    #pragma unroll 2
    for (int it = 0; it < 8; ++it) {
      int gidx = tid + it * 256;
      int m = gidx >> 7, g = gidx & 127;
      int d0 = g * 8;
      float4 f0 = *(const float4*)&sX[m * 1028 + d0];
      float4 f1 = *(const float4*)&sX[m * 1028 + d0 + 4];
      v8bf p;
      p[0] = (__bf16)f0.x; p[1] = (__bf16)f0.y; p[2] = (__bf16)f0.z; p[3] = (__bf16)f0.w;
      p[4] = (__bf16)f1.x; p[5] = (__bf16)f1.y; p[6] = (__bf16)f1.z; p[7] = (__bf16)f1.w;
      int kt = d0 >> 5, kk = d0 & 31;
      int grp = kk >> 3;                 // 0..3 within K-tile
      int khi = grp >> 1, jb = (grp & 1) << 3;
      *(v8bf*)&sA[kt * 512 + (khi * 16 + m) * 16 + jb] = p;
    }
    __syncthreads();

    // --- gate logits: x_l @ G^T, 4 partial threads per (m,e), float4 dots ---
    {
      int p = tid >> 2, chunk = tid & 3;
      int m = p >> 2, e = p & 3;
      const float4* xr = (const float4*)&sX[m * 1028 + chunk * 256];
      const float4* gr = (const float4*)&G[e * D_ + chunk * 256];
      float s = 0.f;
      #pragma unroll 4
      for (int d = 0; d < 64; ++d) {
        float4 a = xr[d], b = gr[d];
        s = fmaf(a.x, b.x, s); s = fmaf(a.y, b.y, s);
        s = fmaf(a.z, b.z, s); s = fmaf(a.w, b.w, s);
      }
      atomicAdd(&sLogit[p], s);
    }
    __syncthreads();

    // --- softmax over E=4 experts (one thread per row) ---
    if (tid < 16) {
      float l0 = sLogit[tid * 4 + 0], l1 = sLogit[tid * 4 + 1];
      float l2 = sLogit[tid * 4 + 2], l3 = sLogit[tid * 4 + 3];
      float mx = fmaxf(fmaxf(l0, l1), fmaxf(l2, l3));
      float e0 = __expf(l0 - mx), e1 = __expf(l1 - mx);
      float e2 = __expf(l2 - mx), e3 = __expf(l3 - mx);
      float inv = 1.f / (e0 + e1 + e2 + e3);
      sGate[tid * 4 + 0] = e0 * inv; sGate[tid * 4 + 1] = e1 * inv;
      sGate[tid * 4 + 2] = e2 * inv; sGate[tid * 4 + 3] = e3 * inv;
    }

    // --- V projection: v1 = tanh(x @ V_e), 2 uniform tasks per wave ---
    #pragma unroll
    for (int rep = 0; rep < 2; ++rep) {
      int task = w + rep * 8;            // uniform (SGPR)
      int e = task >> 2, nt = task & 3;
      int le = l * E_ + e;
      v8f acc = {};
      #pragma unroll 4
      for (int kt = 0; kt < 32; ++kt) {
        v16bf a = *(const v16bf*)&sA[kt * 512 + lane * 16];
        v16bf b = wsV[((le * 4 + nt) * 32 + kt) * 32 + lane];
        acc = __builtin_amdgcn_wmma_f32_16x16x32_bf16(false, a, false, b,
                                                      (short)0, acc, false, false);
      }
      int n = lane & 15, mbase = (lane >> 4) * 8;
      int rg = nt * 16 + n;
      int kt2 = rg >> 5, khi, j;
      a_slot(rg & 31, khi, j);
      #pragma unroll
      for (int i = 0; i < 8; ++i) {
        int m = mbase + i;
        sV1[e * 1024 + kt2 * 512 + (khi * 16 + m) * 16 + j] = (__bf16)fast_tanh(acc[i]);
      }
    }
    __syncthreads();

    // --- C projection: v2 = tanh(v1 @ C_e^T), 2 uniform tasks per wave ---
    #pragma unroll
    for (int rep = 0; rep < 2; ++rep) {
      int task = w + rep * 8;
      int e = task >> 2, nt = task & 3;
      int le = l * E_ + e;
      v8f acc = {};
      #pragma unroll
      for (int kt = 0; kt < 2; ++kt) {
        v16bf a = *(const v16bf*)&sV1[e * 1024 + kt * 512 + lane * 16];
        v16bf b = wsC[((le * 4 + nt) * 2 + kt) * 32 + lane];
        acc = __builtin_amdgcn_wmma_f32_16x16x32_bf16(false, a, false, b,
                                                      (short)0, acc, false, false);
      }
      int n = lane & 15, mbase = (lane >> 4) * 8;
      int rg = nt * 16 + n;
      int kt2 = rg >> 5, khi, j;
      a_slot(rg & 31, khi, j);
      #pragma unroll
      for (int i = 0; i < 8; ++i) {
        int m = mbase + i;
        sV2[e * 1024 + kt2 * 512 + (khi * 16 + m) * 16 + j] = (__bf16)fast_tanh(acc[i]);
      }
    }
    __syncthreads();

    // --- U projection + gated residual epilogue, 8 uniform d-tile tasks per wave ---
    #pragma unroll 2
    for (int rep = 0; rep < 8; ++rep) {
      int dt = w + rep * 8;              // uniform (SGPR)
      v8f acc[E_] = {};
      #pragma unroll
      for (int e = 0; e < E_; ++e) {
        int le = l * E_ + e;
        #pragma unroll
        for (int kt = 0; kt < 2; ++kt) {
          v16bf a = *(const v16bf*)&sV2[e * 1024 + kt * 512 + lane * 16];
          v16bf b = wsU[((le * 64 + dt) * 2 + kt) * 32 + lane];
          acc[e] = __builtin_amdgcn_wmma_f32_16x16x32_bf16(false, a, false, b,
                                                           (short)0, acc[e], false, false);
        }
      }
      int n = lane & 15, mbase = (lane >> 4) * 8;
      int d = dt * 16 + n;
      float bs = bias[l * D_ + d];
      #pragma unroll
      for (int i = 0; i < 8; ++i) {
        int m = mbase + i;
        float s = 0.f;
        #pragma unroll
        for (int e = 0; e < E_; ++e)
          s = fmaf(sGate[m * 4 + e], acc[e][i] + bs, s);
        float xold = sX[m * 1028 + d];
        float xorg = x0g[(row0 + m) * D_ + d];
        sX[m * 1028 + d] = fmaf(xorg, s, xold);
      }
    }
    __syncthreads();
  }

  // Store final residual stream (float4, 16 fixed iters).
  {
    float4* dst = (float4*)(out + row0 * D_);
    #pragma unroll 4
    for (int it = 0; it < 16; ++it) {
      int idx = tid + it * 256;
      int m = idx >> 8, c = idx & 255;
      dst[m * 256 + c] = *(const float4*)&sX[m * 1028 + c * 4];
    }
  }
}

// ---------------------------------------------------------------------------
extern "C" void kernel_launch(void* const* d_in, const int* in_sizes, int n_in,
                              void* d_out, int out_size, void* d_ws, size_t ws_size,
                              hipStream_t stream) {
  (void)in_sizes; (void)n_in; (void)out_size; (void)ws_size;
  const float* inputs = (const float*)d_in[0];
  const float* U      = (const float*)d_in[1];
  const float* V      = (const float*)d_in[2];
  const float* C      = (const float*)d_in[3];
  const float* G      = (const float*)d_in[4];
  const float* bias   = (const float*)d_in[5];
  float*  out = (float*)d_out;
  __bf16* ws  = (__bf16*)d_ws;

  hipLaunchKernelGGL(prep_weights, dim3((WS_TOT + 255) / 256), dim3(256), 0, stream,
                     U, V, C, ws);
  hipLaunchKernelGGL(crossnet_kernel, dim3(B_ / 16), dim3(256), 0, stream,
                     inputs, G, bias, ws, out);
}
